// SparseAttentionOptimizer_65171833749624
// MI455X (gfx1250) — matrix operand order repaired and checked by reading, not compiled, not run on hardware
//
#include <hip/hip_runtime.h>

typedef __attribute__((ext_vector_type(16))) _Float16 v16h;
typedef __attribute__((ext_vector_type(8)))  float    v8f;

#define SEQ_L   4096
#define HEADS   16
#define DH      64
#define NEG_BIG (-1.0e30f)

// One wave (32 threads) handles one (head, 16-query tile).
// Flash-style online softmax over <=20 relevant 16-key tiles:
//   5 local-band tiles  (qbase-32 .. qbase+32)
//   15 strided tiles    (kb = qbase%256 + 256k, skipping the diagonal one)
__global__ __launch_bounds__(32)
void sparse_attn_wmma(const float* __restrict__ Qg,
                      const float* __restrict__ Kg,
                      const float* __restrict__ Vg,
                      float* __restrict__ Og)
{
    __shared__ _Float16 ldsVT[DH * 16];   // V tile transposed: [dim][key], f16
    __shared__ _Float16 ldsP [16 * 16];   // P tile: [row][key], f16

    const int lane  = threadIdx.x;        // 0..31
    const int halfw = lane >> 4;          // 0 | 1
    const int l16   = lane & 15;

    const int h     = blockIdx.x >> 8;         // 16 heads
    const int qbase = (blockIdx.x & 255) << 4; // 256 query tiles of 16

    const size_t hoff = (size_t)h * SEQ_L * DH;
    const float* Qh = Qg + hoff;
    const float* Kh = Kg + hoff;
    const float* Vh = Vg + hoff;
    float*       Oh = Og + hoff;

    // ---- Q A-fragments (16x32 f16 each; two ops cover dims 0..31 / 32..63)
    // A layout (16-bit 16x32): lanes0-15 hold K={0..7,16..23}, lanes16-31 K={8..15,24..31}
    v16h qa0 = {}, qa1 = {};
    {
        const float* qrow = Qh + (size_t)(qbase + l16) * DH;
        const int c0 = halfw * 8;
        #pragma unroll
        for (int e = 0; e < 8; ++e) {
            qa0[e]     = (_Float16)qrow[c0 + e];
            qa0[e + 8] = (_Float16)qrow[c0 + 16 + e];
            qa1[e]     = (_Float16)qrow[c0 + 32 + e];
            qa1[e + 8] = (_Float16)qrow[c0 + 48 + e];
        }
    }

    // Online-softmax state. C/D layout: v8f elem r, lane -> (M = r + 8*halfw, N = l16)
    float m_r[8], l_r[8];
    v8f acc[4];                 // O accumulator, 16 rows x 64 dims (4 dim groups)
    {
        v8f z = {};
        #pragma unroll
        for (int g = 0; g < 4; ++g) acc[g] = z;
        #pragma unroll
        for (int r = 0; r < 8; ++r) { m_r[r] = NEG_BIG; l_r[r] = 0.0f; }
    }

    const float scale = 0.125f;          // 1/sqrt(64)
    const int   qmod  = qbase & 255;

    for (int t = 0; t < 21; ++t) {       // wave-uniform control flow
        int kb;
        if (t < 5) {                     // local band tiles
            kb = qbase + (t - 2) * 16;
            if (kb < 0 || kb >= SEQ_L) continue;
        } else {                         // strided tiles (dedupe diagonal)
            kb = qmod + (t - 5) * 256;
            if (kb == qbase) continue;
        }

        __syncthreads();                 // single wave: cheap; orders LDS reuse

        // ---- K B-fragments (32x16): lane = key column, dims {8h..8h+7, 16+8h..16+8h+7}/lane
        v16h kf0, kf1;
        {
            const float* krow = Kh + (size_t)(kb + l16) * DH + halfw * 8;
            #pragma unroll
            for (int e = 0; e < 16; ++e) {
                kf0[e] = (_Float16)krow[(e & 7) + ((e >> 3) << 4)];        // dims 0..31 across halves
                kf1[e] = (_Float16)krow[32 + (e & 7) + ((e >> 3) << 4)];   // dims 32..63
            }
        }

        // ---- stage V tile transposed into LDS (f16): ldsVT[dim][key]
        {
            const float* vrow = Vh + (size_t)(kb + l16) * DH + halfw * 32;
            #pragma unroll
            for (int e = 0; e < 32; ++e)
                ldsVT[(halfw * 32 + e) * 16 + l16] = (_Float16)vrow[e];
        }

        // ---- scores S = Q K^T (fp32 accum)
        v8f s = {};
        s = __builtin_amdgcn_wmma_f32_16x16x32_f16(false, qa0, false, kf0, (short)0, s, false, false);
        s = __builtin_amdgcn_wmma_f32_16x16x32_f16(false, qa1, false, kf1, (short)0, s, false, false);

        // ---- scale + sparse mask (local band OR stride-256)
        float sv[8];
        #pragma unroll
        for (int r = 0; r < 8; ++r) {
            const int i = qbase + r + 8 * halfw;
            const int d = (kb + l16) - i;
            const bool ok = (d >= -32 && d <= 32) || ((d & 255) == 0);
            sv[r] = ok ? s[r] * scale : NEG_BIG;
        }

        // ---- online softmax: row reductions across the 16-lane half-groups
        float corr[8], p[8];
        #pragma unroll
        for (int r = 0; r < 8; ++r) {
            float mx = sv[r];
            #pragma unroll
            for (int off = 1; off < 16; off <<= 1)
                mx = fmaxf(mx, __shfl_xor(mx, off, 32));
            const float mnew = fmaxf(m_r[r], mx);
            corr[r] = __expf(m_r[r] - mnew);
            p[r]    = __expf(sv[r] - mnew);
            float sum = p[r];
            #pragma unroll
            for (int off = 1; off < 16; off <<= 1)
                sum += __shfl_xor(sum, off, 32);
            l_r[r] = l_r[r] * corr[r] + sum;
            m_r[r] = mnew;
        }

        // ---- transpose P (C layout -> A layout) through LDS
        #pragma unroll
        for (int r = 0; r < 8; ++r)
            ldsP[(r + 8 * halfw) * 16 + l16] = (_Float16)p[r];

        v16h pa = {};                       // elems 8..15 (K=16..31) stay zero
        {
            const _Float16* psrc = &ldsP[l16 * 16 + halfw * 8];
            #pragma unroll
            for (int e = 0; e < 8; ++e) pa[e] = psrc[e];
        }

        // ---- rescale O, then O += P x V  (4 dim groups of 16)
        #pragma unroll
        for (int g = 0; g < 4; ++g) {
            #pragma unroll
            for (int r = 0; r < 8; ++r) acc[g][r] *= corr[r];
            v16h vb;                        // lane = output dim column; keys 8h..8h+7/lane
            const _Float16* vsrc = &ldsVT[(g * 16 + l16) * 16 + halfw * 8];
            #pragma unroll
            for (int e = 0; e < 16; ++e) vb[e] = vsrc[e & 7];  // elems 8..15: dup (x0 in A)
            acc[g] = __builtin_amdgcn_wmma_f32_16x16x32_f16(false, pa, false, vb, (short)0, acc[g], false, false);
        }
    }

    // ---- epilogue: O / l  (coalesced across lanes 0-15 per row)
    #pragma unroll
    for (int g = 0; g < 4; ++g) {
        #pragma unroll
        for (int r = 0; r < 8; ++r) {
            const int row = qbase + r + 8 * halfw;
            const int dim = g * 16 + l16;
            Oh[(size_t)row * DH + dim] = acc[g][r] / l_r[r];
        }
    }
}

extern "C" void kernel_launch(void* const* d_in, const int* in_sizes, int n_in,
                              void* d_out, int out_size, void* d_ws, size_t ws_size,
                              hipStream_t stream) {
    (void)in_sizes; (void)n_in; (void)out_size; (void)d_ws; (void)ws_size;
    const float* Q = (const float*)d_in[0];
    const float* K = (const float*)d_in[1];
    const float* V = (const float*)d_in[2];
    float*       O = (float*)d_out;
    dim3 grid(HEADS * (SEQ_L / 16));   // 4096 waves: one per (head, 16-query tile)
    dim3 block(32);
    sparse_attn_wmma<<<grid, block, 0, stream>>>(Q, K, V, O);
}
